// NaiveSsmLayer_16896401343116
// MI455X (gfx1250) — matrix operand order repaired and checked by reading, not compiled, not run on hardware
//
#include <hip/hip_runtime.h>
#include <hip/hip_bf16.h>

// ---------------------------------------------------------------------------
// SSM layer for MI455X (gfx1250, wave32, WMMA + TDM).
//   y[b,n,o] = C h_n,  h_n = Abar h_{n-1} + Bbar x_n,  Abar/Bbar via bilinear
// - split-precision bf16 WMMA for all f32 setup algebra (log-folded Neumann
//   inverse of (I - dt/2 A), matrix-power doubling)
// - chunked scan: 16 chunks x 128 steps, h resident in LDS, one WMMA sweep
//   per step; per-step Bx tiles (32KB) streamed LDS-side by the Tensor Data
//   Mover, double-buffered, overlapped with the WMMA chain (s_wait_tensorcnt)
// - bulk bf16 WMMA GEMMs for input/output projections
// Memory floor ~330MB @ 23.3 TB/s ~ 14us; FLOPs trivial; serial depth 128+16.
// ---------------------------------------------------------------------------

typedef __attribute__((ext_vector_type(16))) __bf16 v16bf;
typedef __attribute__((ext_vector_type(8)))  __bf16 v8bf;
typedef __attribute__((ext_vector_type(8)))  float  v8f;
typedef __attribute__((ext_vector_type(4)))  unsigned int u32x4;
typedef __attribute__((ext_vector_type(8)))  int i32x8;
typedef __attribute__((ext_vector_type(4)))  int i32x4;

#define BATCH  16
#define SEQN   2048
#define DSTATE 512
#define CHUNK  128
#define NCHUNK (SEQN / CHUNK)      // 16
#define BS     (BATCH * DSTATE)    // 8192 = one h state

__device__ __forceinline__ v8f wmma_bf16(v16bf a, v16bf b, v8f c) {
  // 8 args: (neg_a, A, neg_b, B, c_mod, C, reuse_a, reuse_b)
  return __builtin_amdgcn_wmma_f32_16x16x32_bf16(false, a, false, b,
                                                 (short)0, c, false, false);
}

// A-matrix fragment (16x32 bf16), src row-major with leading dim ld.
// ISA 7.12.2: lane<16 holds row m0+lane, K = {0..7, 16..23};
//             lane>=16 holds row m0+lane-16, K = {8..15, 24..31}.
__device__ __forceinline__ v16bf load_a_frag(const __bf16* base, int ld,
                                             int m0, int k0, int lane) {
  int row  = m0 + (lane & 15);
  int koff = (lane & 16) ? 8 : 0;
  const __bf16* p = base + (size_t)row * ld + k0 + koff;
  v8bf lo = *(const v8bf*)p;
  v8bf hi = *(const v8bf*)(p + 16);
  return __builtin_shufflevector(lo, hi, 0,1,2,3,4,5,6,7,8,9,10,11,12,13,14,15);
}

// B-matrix fragment (32x16 bf16) from "NT" storage: matB[n][k] row-major.
// Lane holds column n0+(lane&15); lanes<16 K={0..7,16..23}, lanes>=16 K={8..15,24..31}.
__device__ __forceinline__ v16bf load_b_frag_nt(const __bf16* base, int ld,
                                                int n0, int k0, int lane) {
  int row  = n0 + (lane & 15);
  int koff = (lane & 16) ? 8 : 0;
  const __bf16* p = base + (size_t)row * ld + k0 + koff;
  v8bf lo = *(const v8bf*)p;
  v8bf hi = *(const v8bf*)(p + 16);
  return __builtin_shufflevector(lo, hi, 0,1,2,3,4,5,6,7,8,9,10,11,12,13,14,15);
}

// ---------------------------------------------------------------------------
// TDM: DMA a contiguous run of `nElemF32` floats from global into LDS.
// D# per ISA ch.8: group0 = {count=1 | lds_addr | global_addr | type=2},
// group1 = {mask=0, data_size=4B, 1-D tile: tile_dim0=nElemF32, huge
// tensor_dim0 so no OOB clipping, stride don't-care}. Groups 2/3 zero (<=2D).
__device__ __forceinline__ unsigned lds_off(const void* p) {
  // LDS aperture: flat address bits [31:0] are the LDS byte offset.
  return (unsigned)(unsigned long long)(size_t)p;
}

__device__ __forceinline__ void tdm_load_to_lds(const void* gsrc, unsigned ldsByteOff,
                                                int nElemF32) {
  unsigned long long ga = (unsigned long long)(size_t)gsrc;
  u32x4 g0 = {0u, 0u, 0u, 0u};
  g0[0] = 1u;                                        // count=1, user descriptor
  g0[1] = ldsByteOff;                                // lds_addr (bytes)
  g0[2] = (unsigned)(ga & 0xffffffffu);              // global_addr[31:0]
  g0[3] = (unsigned)((ga >> 32) & 0x01ffffffu)       // global_addr[56:32]
        | 0x80000000u;                               // type=2 ("image")
  i32x8 g1 = {0, 0, 0, 0, 0, 0, 0, 0};
  g1[0] = 2 << 16;                                   // workgroup_mask=0, data_size=4B
  g1[1] = 0;                                         // no atomic barrier; dim0[15:0]=0
  g1[2] = 0x00010100;                                // tensor_dim0=0x01000000 (16M), tensor_dim1=1
  g1[3] = (nElemF32 & 0xffff) << 16;                 // tile_dim0
  g1[4] = 1;                                         // tile_dim1=1, tile_dim2=0
  g1[5] = nElemF32;                                  // tensor_dim0_stride (don't-care for 1 row)
  i32x4 gz = {0, 0, 0, 0};
#if defined(__clang_major__) && (__clang_major__ >= 23)
  i32x8 gz8 = {0, 0, 0, 0, 0, 0, 0, 0};
  __builtin_amdgcn_tensor_load_to_lds(g0, g1, gz, gz, gz8, 0);
#else
  __builtin_amdgcn_tensor_load_to_lds(g0, g1, gz, gz, 0);
#endif
}

// ---------------------------------------------------------------------------
// Setup: E = 0.5*dt*A, R0 = I + E, dtf = (float)dt
__global__ void k_build(const float* __restrict__ A, const int* __restrict__ dtPtr,
                        float* __restrict__ E, float* __restrict__ R0,
                        float* __restrict__ dtf) {
  float dt = (float)dtPtr[0];
  int i = blockIdx.x * blockDim.x + threadIdx.x;
  if (i == 0) dtf[0] = dt;
  if (i < DSTATE * DSTATE) {
    int r = i >> 9, c = i & (DSTATE - 1);
    float e = 0.5f * dt * A[i];
    E[i]  = e;
    R0[i] = e + (r == c ? 1.0f : 0.0f);
  }
}

__global__ void k_f32_to_bf16(const float* __restrict__ src,
                              __bf16* __restrict__ dst, int n) {
  int i = blockIdx.x * blockDim.x + threadIdx.x;
  int stride = gridDim.x * blockDim.x;
  for (; i < n; i += stride) dst[i] = (__bf16)src[i];
}

// ---------------------------------------------------------------------------
// 512x512x512 f32 GEMM via split-precision bf16 WMMA (hi/lo residual scheme):
//   out = scale * (A*B + (flags&1 ? A : 0))        (A,B,out f32 row-major)
__global__ void __launch_bounds__(256)
k_gemm512(const float* __restrict__ A, const float* __restrict__ Bg,
          float* __restrict__ out, int flags, const float* __restrict__ scalePtr) {
  __shared__ __align__(16) __bf16 Ah[64 * 32], Al[64 * 32];
  __shared__ __align__(16) __bf16 Bh[64 * 32], Bl[64 * 32];  // stored [n][k]
  int tid = threadIdx.x, lane = tid & 31, w = tid >> 5;
  int mBlk = blockIdx.x * 64, nBlk = blockIdx.y * 64;
  v8f zero = {};
  v8f acc[2] = {zero, zero};
  for (int k0 = 0; k0 < 512; k0 += 32) {
    __syncthreads();
    for (int i = tid; i < 64 * 32; i += 256) {
      int r = i >> 5, c = i & 31;
      float va = A[(size_t)(mBlk + r) * 512 + k0 + c];
      __bf16 ha = (__bf16)va;
      Ah[i] = ha;
      Al[i] = (__bf16)(va - (float)ha);
      float vb = Bg[(size_t)(k0 + c) * 512 + nBlk + r];   // transpose into [n][k]
      __bf16 hb = (__bf16)vb;
      Bh[i] = hb;
      Bl[i] = (__bf16)(vb - (float)hb);
    }
    __syncthreads();
    v16bf ah = load_a_frag(Ah, 32, 16 * (w & 3), 0, lane);
    v16bf al = load_a_frag(Al, 32, 16 * (w & 3), 0, lane);
#pragma unroll
    for (int t = 0; t < 2; ++t) {
      int n0 = 16 * ((w >> 2) * 2 + t);
      v16bf bh = load_b_frag_nt(Bh, 32, n0, 0, lane);
      v16bf bl = load_b_frag_nt(Bl, 32, n0, 0, lane);
      acc[t] = wmma_bf16(ah, bh, acc[t]);   // hi*hi
      acc[t] = wmma_bf16(al, bh, acc[t]);   // lo*hi
      acc[t] = wmma_bf16(ah, bl, acc[t]);   // hi*lo
    }
  }
  float scale = scalePtr ? scalePtr[0] : 1.0f;
  int col0 = lane & 15, mOff = (lane & 16) ? 8 : 0;
#pragma unroll
  for (int t = 0; t < 2; ++t) {
    int n = nBlk + 16 * ((w >> 2) * 2 + t) + col0;
#pragma unroll
    for (int r = 0; r < 8; ++r) {
      int m = mBlk + 16 * (w & 3) + mOff + r;
      float v = acc[t][r];
      if (flags & 1) v += A[(size_t)m * 512 + n];
      out[(size_t)m * 512 + n] = scale * v;
    }
  }
}

// ---------------------------------------------------------------------------
// Bulk bf16 GEMM: out[m,n] = sum_k A[m,k]*Bm[n,k]  (both bf16 row-major, "NT").
// mode 0: out row-major [M,N]
// mode 1: m = b*2048+t -> Bx layout [t][b][n]
// mode 2: m = t*16+b   -> y  layout [b][t][n]
__global__ void __launch_bounds__(256)
k_gemm_nt(const __bf16* __restrict__ A, const __bf16* __restrict__ Bm,
          float* __restrict__ out, int M, int Nn, int K, int mode) {
  int tid = threadIdx.x, lane = tid & 31, w = tid >> 5;
  int mBase = blockIdx.x * 64 + 16 * (w & 3);
  int nBase = blockIdx.y * 128 + 64 * (w >> 2);
  v8f zero = {};
  v8f acc[4] = {zero, zero, zero, zero};
  for (int k0 = 0; k0 < K; k0 += 32) {
    v16bf a = load_a_frag(A, K, mBase, k0, lane);
#pragma unroll
    for (int t = 0; t < 4; ++t) {
      v16bf b = load_b_frag_nt(Bm, K, nBase + 16 * t, k0, lane);
      acc[t] = wmma_bf16(a, b, acc[t]);
    }
  }
  int col0 = lane & 15, mOff = (lane & 16) ? 8 : 0;
#pragma unroll
  for (int t = 0; t < 4; ++t) {
#pragma unroll
    for (int r = 0; r < 8; ++r) {
      int m = mBase + mOff + r;
      int n = nBase + 16 * t + col0;
      size_t idx;
      if (mode == 1)
        idx = ((size_t)(m & (SEQN - 1)) * BATCH + (size_t)(m >> 11)) * DSTATE + n;
      else if (mode == 2)
        idx = (size_t)(m & 15) * ((size_t)SEQN * DSTATE) + (size_t)(m >> 4) * DSTATE + n;
      else
        idx = (size_t)m * Nn + n;
      out[idx] = acc[t][r];
    }
  }
}

// ---------------------------------------------------------------------------
// Per-chunk scan: h_t = h_{t-1} * Abar^T + Bx_t, 128 steps, one WGP/chunk.
// h (16x512 bf16) double-buffered in LDS; 32 waves each own 16 columns.
// Bx step tiles (16x512 f32 = 32KB) are TDM-streamed into an LDS double
// buffer one step ahead (wave 0 issues; s_wait_tensorcnt + barrier publish).
__global__ void __launch_bounds__(1024)
k_scan(const __bf16* __restrict__ AbarB, const float* __restrict__ Bx,
       const float* __restrict__ Hinit, float* __restrict__ Ffinal,
       __bf16* __restrict__ hs, int useInit, int storeHs, int storeF) {
  __shared__ __align__(16) __bf16 hbuf[2][BS];
  __shared__ __align__(16) float  bxbuf[2][BS];
  int tid = threadIdx.x, lane = tid & 31, w = tid >> 5;
  int chunk = blockIdx.x;
  int n0 = 16 * w;

  // Warm Abar (bf16, 512KB) toward this WGP's caches.
  for (int i = tid; i < (DSTATE * DSTATE * 2) / 128; i += 1024)
    __builtin_prefetch((const char*)AbarB + (size_t)i * 128, 0, 1);

  // TDM prefetch of step-0 Bx tile while we initialize h.
  if (tid < 32)
    tdm_load_to_lds(Bx + (size_t)(chunk * CHUNK) * BS, lds_off(&bxbuf[0][0]), BS);
  for (int i = tid; i < BS; i += 1024)
    hbuf[0][i] = useInit ? (__bf16)Hinit[(size_t)chunk * BS + i] : (__bf16)0.0f;
  if (tid < 32) __builtin_amdgcn_s_wait_tensorcnt(0);
  __syncthreads();

  int cur = 0, curb = 0;
  int col = n0 + (lane & 15);
  int mOff = (lane & 16) ? 8 : 0;
  for (int j = 0; j < CHUNK; ++j) {
    int t = chunk * CHUNK + j;
    // Issue next step's Bx DMA; it lands in the other buffer while we compute.
    if (j + 1 < CHUNK && tid < 32)
      tdm_load_to_lds(Bx + (size_t)(t + 1) * BS, lds_off(&bxbuf[curb ^ 1][0]), BS);

    v8f acc = {};
    const __bf16* hsrc = &hbuf[cur][0];
#pragma unroll
    for (int k0 = 0; k0 < DSTATE; k0 += 32) {
      v16bf a = load_a_frag(hsrc, DSTATE, 0, k0, lane);
      v16bf b = load_b_frag_nt(AbarB, DSTATE, n0, k0, lane);
      acc = wmma_bf16(a, b, acc);
    }
    const float* bx = &bxbuf[curb][0];
#pragma unroll
    for (int r = 0; r < 8; ++r) acc[r] += bx[(mOff + r) * DSTATE + col];

    __bf16* hdst = &hbuf[cur ^ 1][0];
#pragma unroll
    for (int r = 0; r < 8; ++r) {
      int m = mOff + r;
      __bf16 hv = (__bf16)acc[r];
      hdst[m * DSTATE + col] = hv;
      if (storeHs) hs[(size_t)t * BS + m * DSTATE + col] = hv;
    }
    if (tid < 32) __builtin_amdgcn_s_wait_tensorcnt(0);
    __syncthreads();   // publishes hdst and the TDM'd Bx tile to all waves
    cur ^= 1;
    curb ^= 1;
  }
  if (storeF)
    for (int i = tid; i < BS; i += 1024)
      Ffinal[(size_t)chunk * BS + i] = (float)hbuf[cur][i];
}

// ---------------------------------------------------------------------------
// Chunk-level recurrence: H_c = H_{c-1} * (Abar^128)^T + F_c, 16 steps.
// Writes Hinit[c] = H_{c-1} (shifted) for the phase-3 re-scan; Hinit[0] = 0.
__global__ void __launch_bounds__(1024)
k_chain(const __bf16* __restrict__ ALb, const float* __restrict__ F,
        float* __restrict__ Hinit) {
  __shared__ __align__(16) __bf16 hbuf[BS];
  int tid = threadIdx.x, lane = tid & 31, w = tid >> 5;
  int n0 = 16 * w;
  for (int i = tid; i < BS; i += 1024) {
    Hinit[i] = 0.0f;                 // H_{-1} = 0 for chunk 0
    float h0 = F[i];                 // H_0 = F_0
    hbuf[i] = (__bf16)h0;
    Hinit[BS + i] = h0;              // start state for chunk 1
  }
  __syncthreads();
  int col = n0 + (lane & 15);
  int mOff = (lane & 16) ? 8 : 0;
  for (int c = 1; c < NCHUNK; ++c) {
    v8f acc = {};
#pragma unroll
    for (int k0 = 0; k0 < DSTATE; k0 += 32) {
      v16bf a = load_a_frag(hbuf, DSTATE, 0, k0, lane);
      v16bf b = load_b_frag_nt(ALb, DSTATE, n0, k0, lane);
      acc = wmma_bf16(a, b, acc);
    }
    float vals[8];
#pragma unroll
    for (int r = 0; r < 8; ++r)
      vals[r] = acc[r] + F[(size_t)c * BS + (mOff + r) * DSTATE + col];
    __syncthreads();   // everyone done reading hbuf before overwrite
#pragma unroll
    for (int r = 0; r < 8; ++r) {
      int m = mOff + r;
      hbuf[m * DSTATE + col] = (__bf16)vals[r];
      if (c < NCHUNK - 1)
        Hinit[(size_t)(c + 1) * BS + m * DSTATE + col] = vals[r];
    }
    __syncthreads();
  }
}

// ---------------------------------------------------------------------------
extern "C" void kernel_launch(void* const* d_in, const int* in_sizes, int n_in,
                              void* d_out, int out_size, void* d_ws, size_t ws_size,
                              hipStream_t stream) {
  (void)in_sizes; (void)n_in; (void)out_size; (void)ws_size;
  const float* signal = (const float*)d_in[0];   // [16,2048,512] f32
  const float* Amat   = (const float*)d_in[1];   // [512,512]
  const float* Bmat   = (const float*)d_in[2];   // [512,512]
  const float* Cmat   = (const float*)d_in[3];   // [512,512]
  const int*   dtPtr  = (const int*)d_in[4];     // scalar 1

  char* ws = (char*)d_ws;
  const size_t MB = 1u << 20;
  float*  E     = (float*)(ws + 0 * MB);
  float*  R1    = (float*)(ws + 1 * MB);
  float*  R2    = (float*)(ws + 2 * MB);
  float*  S1    = (float*)(ws + 3 * MB);
  float*  S2    = (float*)(ws + 4 * MB);
  float*  Abar  = (float*)(ws + 5 * MB);
  float*  Bbar  = (float*)(ws + 6 * MB);
  float*  P1    = (float*)(ws + 7 * MB);
  float*  P2    = (float*)(ws + 8 * MB);
  float*  dtf   = (float*)(ws + 9 * MB);
  float*  F     = (float*)(ws + 10 * MB);                 // 512KB
  float*  Hinit = (float*)(ws + 10 * MB + 512 * 1024);    // 512KB
  __bf16* AbarB = (__bf16*)(ws + 11 * MB);                // 512KB, [n][k]
  __bf16* ALb   = (__bf16*)(ws + 11 * MB + 512 * 1024);   // 512KB
  __bf16* BbarB = (__bf16*)(ws + 12 * MB);                // 512KB, [s][i]
  __bf16* Cb    = (__bf16*)(ws + 12 * MB + 512 * 1024);   // 512KB, [o][s]
  __bf16* sigB  = (__bf16*)(ws + 13 * MB);                // 32MB, [b*2048+t][i]
  __bf16* hsB   = (__bf16*)(ws + 45 * MB);                // 32MB, [t*16+b][s]
  float*  Bx    = (float*)(ws + 77 * MB);                 // 64MB, [t][b][s]

  dim3 g64(8, 8);     // 512x512 tiles of 64
  dim3 gBig(512, 4);  // 32768x512, tiles 64x128

  // --- discretization: E = 0.5*dt*A, R = Pi(I + E^{2^k}) ~= (I-E)^{-1}
  k_build<<<1024, 256, 0, stream>>>(Amat, dtPtr, E, R1, dtf);
  float *Rc = R1, *Rn = R2, *Sc = S1, *Sn = S2;
  k_gemm512<<<g64, 256, 0, stream>>>(E, E, Sc, 0, nullptr);        // S = E^2
  for (int k = 0; k < 7; ++k) {                                    // fold E^2..E^128
    k_gemm512<<<g64, 256, 0, stream>>>(Rc, Sc, Rn, 1, nullptr);    // R = R*(I+S)
    { float* tmp = Rc; Rc = Rn; Rn = tmp; }
    if (k < 6) {
      k_gemm512<<<g64, 256, 0, stream>>>(Sc, Sc, Sn, 0, nullptr);  // S = S^2
      { float* tmp = Sc; Sc = Sn; Sn = tmp; }
    }
  }
  // Abar = M^{-1}(I+E) = R*(I+E);  Bbar = dt * M^{-1} * B
  k_gemm512<<<g64, 256, 0, stream>>>(Rc, E, Abar, 1, nullptr);
  k_gemm512<<<g64, 256, 0, stream>>>(Rc, Bmat, Bbar, 0, dtf);
  // Abar^128 via 7 squarings (ends in P1)
  k_gemm512<<<g64, 256, 0, stream>>>(Abar, Abar, P1, 0, nullptr);
  k_gemm512<<<g64, 256, 0, stream>>>(P1, P1, P2, 0, nullptr);
  k_gemm512<<<g64, 256, 0, stream>>>(P2, P2, P1, 0, nullptr);
  k_gemm512<<<g64, 256, 0, stream>>>(P1, P1, P2, 0, nullptr);
  k_gemm512<<<g64, 256, 0, stream>>>(P2, P2, P1, 0, nullptr);
  k_gemm512<<<g64, 256, 0, stream>>>(P1, P1, P2, 0, nullptr);
  k_gemm512<<<g64, 256, 0, stream>>>(P2, P2, P1, 0, nullptr);

  // --- bf16 operand prep
  k_f32_to_bf16<<<512, 256, 0, stream>>>(Abar, AbarB, DSTATE * DSTATE);
  k_f32_to_bf16<<<512, 256, 0, stream>>>(P1, ALb, DSTATE * DSTATE);
  k_f32_to_bf16<<<512, 256, 0, stream>>>(Bbar, BbarB, DSTATE * DSTATE);
  k_f32_to_bf16<<<512, 256, 0, stream>>>(Cmat, Cb, DSTATE * DSTATE);
  k_f32_to_bf16<<<4096, 256, 0, stream>>>(signal, sigB, BATCH * SEQN * DSTATE);

  // --- Bx[t][b][s] = sum_i signal[b,t,i] * Bbar[s,i]
  k_gemm_nt<<<gBig, 256, 0, stream>>>(sigB, BbarB, Bx, BATCH * SEQN, DSTATE, DSTATE, 1);

  // --- chunked scan: local pass -> chunk chain -> corrected pass
  k_scan<<<NCHUNK, 1024, 0, stream>>>(AbarB, Bx, Hinit, F, hsB, 0, 0, 1);
  k_chain<<<1, 1024, 0, stream>>>(ALb, F, Hinit);
  k_scan<<<NCHUNK, 1024, 0, stream>>>(AbarB, Bx, Hinit, F, hsB, 1, 1, 0);

  // --- y[b][t][o] = sum_s hs[t,b,s] * C[o,s]
  k_gemm_nt<<<gBig, 256, 0, stream>>>(hsB, Cb, (float*)d_out, BATCH * SEQN, DSTATE, DSTATE, 2);
}
